// MultiHeadAttentionLSA_13657996001585
// MI455X (gfx1250) — matrix operand order repaired and checked by reading, not compiled, not run on hardware
//
#include <hip/hip_runtime.h>
#include <hip/hip_bf16.h>
#include <stdint.h>

#define EMBED   768
#define NHEAD   12
#define HDIM    64
#define BATCH   8
#define SEQ     1024
#define ROWS    (BATCH*SEQ)     /* 8192 */
#define QKV_OUT (3*EMBED)       /* 2304 */

typedef __attribute__((ext_vector_type(16))) __bf16 v16bf;
typedef __attribute__((ext_vector_type(8)))  float  v8f;

union Frag16 {
    v16bf v;
    uint4 q[2];
};

#define WMMA_BF16(A, B, C) \
    __builtin_amdgcn_wmma_f32_16x16x32_bf16(false, (A), false, (B), (short)0, (C), false, false)

// ---------------------------------------------------------------------------
// A-matrix fragment (16x32 bf16), row-major source, ld elements (mult of 8).
// src points at (m0, k0). ISA layout: lanes0-15 = M, V0..3:K=kh*8+0..7,
// V4..7:K=16+kh*8+0..7  -> two contiguous 16B chunks per lane.
// ---------------------------------------------------------------------------
__device__ __forceinline__ v16bf load_a_frag(const __bf16* src, int ld) {
    int lane = threadIdx.x & 31;
    int m = lane & 15, kh = lane >> 4;
    const uint4* p = (const uint4*)(src + (size_t)m * ld);
    Frag16 f;
    f.q[0] = p[kh];       // bf16 K = kh*8 .. kh*8+7
    f.q[1] = p[kh + 2];   // bf16 K = 16+kh*8 .. 16+kh*8+7
    return f.v;
}

// ---------------------------------------------------------------------------
// B-matrix fragment (32x16 bf16) where B[k][n] = src[n*ld + k] ("n-major").
// src points at (n0, k0), ld mult of 8. N=lane&15, kh=lane>>4: elems 0..7 K=kh*8+0..7,
// elems 8..15 K=16+kh*8+0..7 (matches A layout).
// ---------------------------------------------------------------------------
__device__ __forceinline__ v16bf load_b_frag(const __bf16* src, int ld) {
    int lane = threadIdx.x & 31;
    int n = lane & 15, kh = lane >> 4;
    const uint4* p = (const uint4*)(src + (size_t)n * ld);
    Frag16 f;
    f.q[0] = p[kh];
    f.q[1] = p[kh + 2];
    return f.v;
}

// ---------------------------------------------------------------------------
// fp32 -> bf16 conversion, 4 elements / thread (all sizes are multiples of 4)
// ---------------------------------------------------------------------------
__global__ void cvt_f32_bf16(const float4* __restrict__ src,
                             __bf16* __restrict__ dst, int n4) {
    int i = blockIdx.x * blockDim.x + threadIdx.x;
    if (i < n4) {
        float4 f = src[i];
        union { __bf16 b[4]; uint2 u; } pk;
        pk.b[0] = (__bf16)f.x; pk.b[1] = (__bf16)f.y;
        pk.b[2] = (__bf16)f.z; pk.b[3] = (__bf16)f.w;
        *(uint2*)(dst + 4 * (size_t)i) = pk.u;
    }
}

// ---------------------------------------------------------------------------
// Pipelined 32x64 wave-tile GEMM core: C += A(32xK) * B(Kx64)^T with
// A row-major [m, k] (ld=EMBED) and B "n-major" [n, k] (ld=EMBED).
// Fragments for step k+32 are loaded into fresh registers before the 8 WMMAs
// of step k, so loads overlap matrix ops (partial loadcnt waits).
// ---------------------------------------------------------------------------
__device__ __forceinline__ void gemm32x64(const __bf16* __restrict__ aptr,
                                          const __bf16* __restrict__ bptr,
                                          v8f acc[8]) {
    v16bf a0 = load_a_frag(aptr, EMBED);
    v16bf a1 = load_a_frag(aptr + (size_t)16 * EMBED, EMBED);
    v16bf b0 = load_b_frag(bptr, EMBED);
    v16bf b1 = load_b_frag(bptr + (size_t)16 * EMBED, EMBED);
    v16bf b2 = load_b_frag(bptr + (size_t)32 * EMBED, EMBED);
    v16bf b3 = load_b_frag(bptr + (size_t)48 * EMBED, EMBED);
    for (int k0 = 0; k0 < EMBED; k0 += 32) {
        int kn = (k0 + 32 < EMBED) ? (k0 + 32) : 0;  // wrapped prefetch
        v16bf na0 = load_a_frag(aptr + kn, EMBED);
        v16bf na1 = load_a_frag(aptr + (size_t)16 * EMBED + kn, EMBED);
        v16bf nb0 = load_b_frag(bptr + kn, EMBED);
        v16bf nb1 = load_b_frag(bptr + (size_t)16 * EMBED + kn, EMBED);
        v16bf nb2 = load_b_frag(bptr + (size_t)32 * EMBED + kn, EMBED);
        v16bf nb3 = load_b_frag(bptr + (size_t)48 * EMBED + kn, EMBED);
        acc[0] = WMMA_BF16(a0, b0, acc[0]);
        acc[1] = WMMA_BF16(a0, b1, acc[1]);
        acc[2] = WMMA_BF16(a0, b2, acc[2]);
        acc[3] = WMMA_BF16(a0, b3, acc[3]);
        acc[4] = WMMA_BF16(a1, b0, acc[4]);
        acc[5] = WMMA_BF16(a1, b1, acc[5]);
        acc[6] = WMMA_BF16(a1, b2, acc[6]);
        acc[7] = WMMA_BF16(a1, b3, acc[7]);
        a0 = na0; a1 = na1;
        b0 = nb0; b1 = nb1; b2 = nb2; b3 = nb3;
    }
}

// ---------------------------------------------------------------------------
// QKV GEMM: y[r,o] = sum_c xb[r,c]*wb[o,c]; scatter to per-head Q/K (n,d) and
// transposed V (d,n). Block = 8 waves as 2(M)x4(N): block tile 64x256.
// ---------------------------------------------------------------------------
__global__ __launch_bounds__(256) void qkv_gemm(
        const __bf16* __restrict__ xb, const __bf16* __restrict__ wb,
        __bf16* __restrict__ qb, __bf16* __restrict__ kb,
        __bf16* __restrict__ vb) {
    int wave = threadIdx.x >> 5, lane = threadIdx.x & 31;
    int mw = wave & 1, nw = wave >> 1;
    int m0 = blockIdx.x * 64 + mw * 32;    // gridDim.x = 128
    int n0 = blockIdx.y * 256 + nw * 64;   // gridDim.y = 9

    v8f acc[8] = {};
    gemm32x64(xb + (size_t)m0 * EMBED, wb + (size_t)n0 * EMBED, acc);

    // n0 is 64-aligned and 768 is a multiple of 64 => which/h uniform per wave
    int which = n0 / EMBED;
    int h     = (n0 % EMBED) / HDIM;
    int col = lane & 15, rowsel = lane >> 4;
    if (which < 2) {
        __bf16* dst = (which == 0) ? qb : kb;
#pragma unroll
        for (int mi = 0; mi < 2; ++mi)
#pragma unroll
            for (int t = 0; t < 4; ++t) {
                int d = t * 16 + col;
#pragma unroll
                for (int v = 0; v < 8; ++v) {
                    int r = m0 + mi * 16 + v + 8 * rowsel;
                    int b = r >> 10, n = r & 1023;
                    dst[(((size_t)b * NHEAD + h) * SEQ + n) * HDIM + d] =
                        (__bf16)acc[mi * 4 + t][v];
                }
            }
    } else {
#pragma unroll
        for (int mi = 0; mi < 2; ++mi)
#pragma unroll
            for (int t = 0; t < 4; ++t) {
                int d = t * 16 + col;
#pragma unroll
                for (int v = 0; v < 8; ++v) {
                    int r = m0 + mi * 16 + v + 8 * rowsel;
                    int b = r >> 10, n = r & 1023;
                    vb[(((size_t)b * NHEAD + h) * HDIM + d) * SEQ + n] =
                        (__bf16)acc[mi * 4 + t][v];
                }
            }
    }
}

// ---------------------------------------------------------------------------
// Flash attention: one wave per 16 query rows of one (b,h).
// Reference's mean-subtraction before softmax is a no-op (shift invariance),
// so the exact result equals a standard max-subtracted online softmax.
// K-fragments are double-buffered across key blocks; V-fragments are issued
// before the S WMMAs so the softmax VALU section hides their latency.
// ---------------------------------------------------------------------------
__global__ __launch_bounds__(256) void attn_kernel(
        const __bf16* __restrict__ qb, const __bf16* __restrict__ kb,
        const __bf16* __restrict__ vb, const float* __restrict__ temperature,
        __bf16* __restrict__ ob) {
    __shared__ __align__(16) __bf16 pbuf[8][16 * 32];   // 1KB per wave

    int wave = threadIdx.x >> 5, lane = threadIdx.x & 31;
    int task = blockIdx.x * 8 + wave;      // gridDim.x = 768 -> 6144 tasks
    int bh = task >> 6;                    // (b*H+h)
    int qt = task & 63;
    int h = bh % NHEAD;
    int qm0 = qt * 16;
    float factor = temperature[h] * 0.125f;  // D^-0.5 = 1/8

    const __bf16* Q = qb + (size_t)bh * SEQ * HDIM + (size_t)qm0 * HDIM;
    const __bf16* K = kb + (size_t)bh * SEQ * HDIM;
    const __bf16* V = vb + (size_t)bh * HDIM * SEQ;

    v16bf aq0 = load_a_frag(Q, HDIM);        // d = 0..31
    v16bf aq1 = load_a_frag(Q + 32, HDIM);   // d = 32..63

    v8f acc0 = {}, acc1 = {}, acc2 = {}, acc3 = {};
    float mrow[8], lrow[8];
#pragma unroll
    for (int v = 0; v < 8; ++v) { mrow[v] = -1e30f; lrow[v] = 0.f; }

    int col = lane & 15, rowsel = lane >> 4;
    __bf16* myp = &pbuf[wave][0];

    // Pre-load K fragments for the first key block
    v16bf kf0 = load_b_frag(K, HDIM);
    v16bf kf1 = load_b_frag(K + 32, HDIM);
    v16bf kf2 = load_b_frag(K + (size_t)16 * HDIM, HDIM);
    v16bf kf3 = load_b_frag(K + (size_t)16 * HDIM + 32, HDIM);

    for (int m0 = 0; m0 < SEQ; m0 += 32) {
        // Issue V fragment loads early; consumed after softmax
        v16bf vf0 = load_b_frag(V + m0, SEQ);
        v16bf vf1 = load_b_frag(V + (size_t)16 * SEQ + m0, SEQ);
        v16bf vf2 = load_b_frag(V + (size_t)32 * SEQ + m0, SEQ);
        v16bf vf3 = load_b_frag(V + (size_t)48 * SEQ + m0, SEQ);

        // S tile 16x32: two 16x16 C frags, K=64 via two WMMA k-steps each
        v8f s0 = {}, s1 = {};
        s0 = WMMA_BF16(aq0, kf0, s0);
        s0 = WMMA_BF16(aq1, kf1, s0);
        s1 = WMMA_BF16(aq0, kf2, s1);
        s1 = WMMA_BF16(aq1, kf3, s1);

        // Double-buffer: K fragments for the next key block (wrapped)
        int mn = (m0 + 32 < SEQ) ? (m0 + 32) : 0;
        kf0 = load_b_frag(K + (size_t)mn * HDIM, HDIM);
        kf1 = load_b_frag(K + (size_t)mn * HDIM + 32, HDIM);
        kf2 = load_b_frag(K + (size_t)(mn + 16) * HDIM, HDIM);
        kf3 = load_b_frag(K + (size_t)(mn + 16) * HDIM + 32, HDIM);

        // Online softmax update per row (row = VGPR index + 8*rowsel);
        // a row's 16 values live across the 16 lanes of a half-wave.
#pragma unroll
        for (int v = 0; v < 8; ++v) {
            float a0 = s0[v] * factor, a1 = s1[v] * factor;
            float mx = fmaxf(a0, a1);
            mx = fmaxf(mx, __shfl_xor(mx, 1));
            mx = fmaxf(mx, __shfl_xor(mx, 2));
            mx = fmaxf(mx, __shfl_xor(mx, 4));
            mx = fmaxf(mx, __shfl_xor(mx, 8));
            float newm = fmaxf(mrow[v], mx);
            float alpha = __expf(mrow[v] - newm);
            float p0 = __expf(a0 - newm), p1 = __expf(a1 - newm);
            float ps = p0 + p1;
            ps += __shfl_xor(ps, 1);
            ps += __shfl_xor(ps, 2);
            ps += __shfl_xor(ps, 4);
            ps += __shfl_xor(ps, 8);
            lrow[v] = lrow[v] * alpha + ps;
            mrow[v] = newm;
            acc0[v] *= alpha; acc1[v] *= alpha;
            acc2[v] *= alpha; acc3[v] *= alpha;
            int row = v + 8 * rowsel;
            myp[row * 32 + col]      = (__bf16)p0;
            myp[row * 32 + 16 + col] = (__bf16)p1;
        }
        // C-layout -> A-layout transpose of P through this wave's LDS slab
        asm volatile("s_wait_dscnt 0" ::: "memory");
        v16bf pa = load_a_frag(myp, 32);   // P 16x32 (K = 32 keys)

        // P @ V : V stored transposed [d][n] -> contiguous per-lane B loads
        acc0 = WMMA_BF16(pa, vf0, acc0);
        acc1 = WMMA_BF16(pa, vf1, acc1);
        acc2 = WMMA_BF16(pa, vf2, acc2);
        acc3 = WMMA_BF16(pa, vf3, acc3);
    }

    // Normalize and store to ob in [B,N,C] layout (bf16) for the final GEMM
    int b = bh / NHEAD;
#pragma unroll
    for (int v = 0; v < 8; ++v) {
        float inv = 1.0f / lrow[v];
        int n = qm0 + v + 8 * rowsel;
        size_t base = ((size_t)b * SEQ + n) * EMBED + (size_t)h * HDIM + col;
        ob[base +  0] = (__bf16)(acc0[v] * inv);
        ob[base + 16] = (__bf16)(acc1[v] * inv);
        ob[base + 32] = (__bf16)(acc2[v] * inv);
        ob[base + 48] = (__bf16)(acc3[v] * inv);
    }
}

// ---------------------------------------------------------------------------
// Output projection: out[r,o] = sum_c ob[r,c]*wproj[o,c] + bias[o] (fp32 out)
// ---------------------------------------------------------------------------
__global__ __launch_bounds__(256) void proj_gemm(
        const __bf16* __restrict__ ob, const __bf16* __restrict__ wb,
        const float* __restrict__ bias, float* __restrict__ out) {
    int wave = threadIdx.x >> 5, lane = threadIdx.x & 31;
    int mw = wave & 1, nw = wave >> 1;
    int m0 = blockIdx.x * 64 + mw * 32;    // gridDim.x = 128
    int n0 = blockIdx.y * 256 + nw * 64;   // gridDim.y = 3

    v8f acc[8] = {};
    gemm32x64(ob + (size_t)m0 * EMBED, wb + (size_t)n0 * EMBED, acc);

    int col = lane & 15, rowsel = lane >> 4;
#pragma unroll
    for (int mi = 0; mi < 2; ++mi)
#pragma unroll
        for (int t = 0; t < 4; ++t) {
            int o = n0 + t * 16 + col;
            float bv = bias[o];
#pragma unroll
            for (int v = 0; v < 8; ++v) {
                int r = m0 + mi * 16 + v + 8 * rowsel;
                out[(size_t)r * EMBED + o] = acc[mi * 4 + t][v] + bv;
            }
        }
}

// ---------------------------------------------------------------------------
extern "C" void kernel_launch(void* const* d_in, const int* in_sizes, int n_in,
                              void* d_out, int out_size, void* d_ws,
                              size_t ws_size, hipStream_t stream) {
    (void)in_sizes; (void)n_in; (void)out_size; (void)ws_size;
    const float* x      = (const float*)d_in[0];
    const float* w_qkv  = (const float*)d_in[1];
    const float* w_proj = (const float*)d_in[2];
    const float* b_proj = (const float*)d_in[3];
    const float* temp   = (const float*)d_in[4];

    char* ws = (char*)d_ws;
    size_t off = 0;
    auto alloc = [&](size_t bytes) -> void* {
        void* p = ws + off;
        off += (bytes + 255) & ~(size_t)255;
        return p;
    };
    __bf16* xb     = (__bf16*)alloc((size_t)ROWS * EMBED * 2);
    __bf16* wqkvb  = (__bf16*)alloc((size_t)QKV_OUT * EMBED * 2);
    __bf16* wprojb = (__bf16*)alloc((size_t)EMBED * EMBED * 2);
    __bf16* qb     = (__bf16*)alloc((size_t)BATCH * NHEAD * SEQ * HDIM * 2);
    __bf16* kb     = (__bf16*)alloc((size_t)BATCH * NHEAD * SEQ * HDIM * 2);
    __bf16* vb     = (__bf16*)alloc((size_t)BATCH * NHEAD * SEQ * HDIM * 2);
    __bf16* ob     = (__bf16*)alloc((size_t)ROWS * EMBED * 2);

    int nx = ROWS * EMBED, nw1 = QKV_OUT * EMBED, nw2 = EMBED * EMBED;
    cvt_f32_bf16<<<(nx / 4 + 255) / 256, 256, 0, stream>>>(
        (const float4*)x, xb, nx / 4);
    cvt_f32_bf16<<<(nw1 / 4 + 255) / 256, 256, 0, stream>>>(
        (const float4*)w_qkv, wqkvb, nw1 / 4);
    cvt_f32_bf16<<<(nw2 / 4 + 255) / 256, 256, 0, stream>>>(
        (const float4*)w_proj, wprojb, nw2 / 4);

    qkv_gemm<<<dim3(ROWS / 64, QKV_OUT / 256), 256, 0, stream>>>(
        xb, wqkvb, qb, kb, vb);

    attn_kernel<<<(BATCH * NHEAD * (SEQ / 16)) / 8, 256, 0, stream>>>(
        qb, kb, vb, temp, ob);

    proj_gemm<<<dim3(ROWS / 64, EMBED / 256), 256, 0, stream>>>(
        ob, wprojb, b_proj, (float*)d_out);
}
